// GAT_86912958202581
// MI455X (gfx1250) — matrix operand order repaired and checked by reading, not compiled, not run on hardware
//
#include <hip/hip_runtime.h>
#include <hip/hip_bf16.h>
#include <cmath>

// ---------------------------------------------------------------------------
// GAT forward for gfx1250 (MI455X).
// Dense GEMMs: v_wmma_f32_16x16x32_bf16, wave32. Block = 4 waves = 64 rows.
// The shared 64-col W tile is converted to bf16 once per block into LDS
// (<=24KB of the 320KB WGP LDS); each wave then feeds 4 accumulators
// (16x64 strip) from ds_load_b128 B-fragments + global A-fragments.
// K / NOUT are template constants -> fully unrolled, immediate offsets.
// Edge softmax/aggregation: bandwidth-bound, one wave per (edge, head).
// ---------------------------------------------------------------------------

typedef __attribute__((ext_vector_type(16))) __bf16 v16bf;
typedef __attribute__((ext_vector_type(8)))  float  v8f;

// A fragment: 16x32 bf16. lanes 0-15: row M=lane, K = {k0..k0+7, k0+16..k0+23}
//             lanes 16-31: row M=lane-16, K = {k0+8..k0+15, k0+24..k0+31}
__device__ __forceinline__ v16bf load_a_frag(const float* __restrict__ A,
                                             int lda, int row, int k0, int hi) {
  const float* p = A + (size_t)row * lda + k0 + hi * 8;
  v16bf a;
#pragma unroll
  for (int i = 0; i < 8; ++i) a[i] = (__bf16)p[i];
#pragma unroll
  for (int i = 0; i < 8; ++i) a[8 + i] = (__bf16)p[16 + i];
  return a;
}

// C = A[M,K] @ W[K,NOUT], row-major f32. One wave -> 16 rows x 64 cols.
// Block = 4 waves = 64 rows. Requires M % 16 == 0 and NOUT % 64 == 0.
template <int K, int NOUT>
__global__ __launch_bounds__(128, 1)
void gemm_wmma_bf16(const float* __restrict__ A,
                    const float* __restrict__ W,
                    float* __restrict__ C, int M) {
  // LDS-staged B tile: bf16 [64][K], n-major (transposed).
  __shared__ __align__(16) __bf16 Bs[K * 64];

  const int tid = threadIdx.x;
  const int lane = tid & 31;
  const int wave = tid >> 5;
  const int n0 = blockIdx.y * 64;

  // Cooperative convert W[:, n0:n0+64] -> LDS bf16 (pairs per thread).
#pragma unroll 2
  for (int idx = tid * 2; idx < K * 64; idx += 256) {
    const int kk = idx >> 6;
    const int nn = idx & 63;
    const float* wp = W + (size_t)kk * NOUT + n0 + nn;
    Bs[nn * K + kk]       = (__bf16)wp[0];
    Bs[(nn + 1) * K + kk] = (__bf16)wp[1];
  }
  __syncthreads();

  const int m0 = (blockIdx.x * 4 + wave) * 16;
  if (m0 >= M) return;                  // uniform per wave: EXEC stays all-1s
  const int hi = lane >> 4;
  const int rowA = m0 + (lane & 15);

  v8f acc0 = {}, acc1 = {}, acc2 = {}, acc3 = {};
#pragma unroll
  for (int k0 = 0; k0 < K; k0 += 32) {
    v16bf a = load_a_frag(A, K, rowA, k0, hi);
    // B fragment for sub-tile t: lane = N col (16t+lane%16), K in A's table order.
    const __bf16* bcol = Bs + (size_t)(lane & 15) * K + k0 + hi * 8;
    v16bf b0, b1, b2, b3;
#pragma unroll
    for (int i = 0; i < 8; ++i) {
      b0[i] = bcol[i];               b0[8 + i] = bcol[16 + i];
      b1[i] = bcol[16 * K + i];      b1[8 + i] = bcol[16 * K + 16 + i];
      b2[i] = bcol[32 * K + i];      b2[8 + i] = bcol[32 * K + 16 + i];
      b3[i] = bcol[48 * K + i];      b3[8 + i] = bcol[48 * K + 16 + i];
    }
    acc0 = __builtin_amdgcn_wmma_f32_16x16x32_bf16(false, a, false, b0, (short)0, acc0, false, false);
    acc1 = __builtin_amdgcn_wmma_f32_16x16x32_bf16(false, a, false, b1, (short)0, acc1, false, false);
    acc2 = __builtin_amdgcn_wmma_f32_16x16x32_bf16(false, a, false, b2, (short)0, acc2, false, false);
    acc3 = __builtin_amdgcn_wmma_f32_16x16x32_bf16(false, a, false, b3, (short)0, acc3, false, false);
  }

  // C/D layout: VGPR r -> row m0 + 8*hi + r, col = n0 + 16*t + lane%16.
  float* cp = C + (size_t)(m0 + hi * 8) * NOUT + n0 + (lane & 15);
#pragma unroll
  for (int r = 0; r < 8; ++r) {
    cp[(size_t)r * NOUT +  0] = acc0[r];
    cp[(size_t)r * NOUT + 16] = acc1[r];
    cp[(size_t)r * NOUT + 32] = acc2[r];
    cp[(size_t)r * NOUT + 48] = acc3[r];
  }
}

// ---------------- elementwise / fill helpers ----------------

__global__ void fill_f32(float* __restrict__ p, float v, long long n) {
  long long i = (long long)blockIdx.x * blockDim.x + threadIdx.x;
  long long st = (long long)gridDim.x * blockDim.x;
  for (; i < n; i += st) p[i] = v;
}

// alpha_{src,dst}[n,h] = sum_c h[n,h,c] * a_{src,dst}[h,c]
template <int HEADS>
__global__ void alpha_reduce(const float* __restrict__ h,
                             const float* __restrict__ a_src,
                             const float* __restrict__ a_dst,
                             float* __restrict__ as, float* __restrict__ ad,
                             int Nn) {
  long long i = (long long)blockIdx.x * blockDim.x + threadIdx.x;
  if (i >= (long long)Nn * HEADS) return;
  int n = (int)(i / HEADS), hd = (int)(i % HEADS);
  const float* hp = h + ((size_t)n * HEADS + hd) * 64;
  const float* sp = a_src + hd * 64;
  const float* dp = a_dst + hd * 64;
  float s1 = 0.f, s2 = 0.f;
#pragma unroll 8
  for (int c = 0; c < 64; ++c) { float v = hp[c]; s1 += v * sp[c]; s2 += v * dp[c]; }
  as[i] = s1; ad[i] = s2;
}

__device__ __forceinline__ void atomicMaxF32(float* addr, float val) {
  if (val >= 0.0f) atomicMax((int*)addr, __float_as_int(val));
  else             atomicMin((unsigned int*)addr, __float_as_uint(val));
}

__device__ __forceinline__ float leaky02(float v) { return v > 0.f ? v : 0.2f * v; }

// pass 1: segment max of leaky_relu(as[src]+ad[dst]) over dst
template <int HEADS>
__global__ void edge_max_k(const float* __restrict__ as, const float* __restrict__ ad,
                           const int* __restrict__ src, const int* __restrict__ dst,
                           float* __restrict__ emax, long long total) {
  long long i = (long long)blockIdx.x * blockDim.x + threadIdx.x;
  if (i >= total) return;
  int e = (int)(i / HEADS), hd = (int)(i % HEADS);
  int s = src[e], d = dst[e];
  float v = leaky02(as[(size_t)s * HEADS + hd] + ad[(size_t)d * HEADS + hd]);
  atomicMaxF32(&emax[(size_t)d * HEADS + hd], v);
}

// pass 2: ex = exp(e - emax[dst]); denom[dst] += ex
template <int HEADS>
__global__ void edge_exp_k(const float* __restrict__ as, const float* __restrict__ ad,
                           const int* __restrict__ src, const int* __restrict__ dst,
                           const float* __restrict__ emax,
                           float* __restrict__ EX, float* __restrict__ DEN,
                           long long total) {
  long long i = (long long)blockIdx.x * blockDim.x + threadIdx.x;
  if (i >= total) return;
  int e = (int)(i / HEADS), hd = (int)(i % HEADS);
  int s = src[e], d = dst[e];
  float v = leaky02(as[(size_t)s * HEADS + hd] + ad[(size_t)d * HEADS + hd]);
  float ex = expf(v - emax[(size_t)d * HEADS + hd]);
  EX[i] = ex;
  atomicAdd(&DEN[(size_t)d * HEADS + hd], ex);
}

// pass 3: out[dst] += (ex/denom[dst]) * h[src]; one wave per (edge,head),
// lanes cover 64 channels (2 per lane) -> coalesced gather + atomic scatter.
template <int HEADS>
__global__ void edge_agg_k(const float* __restrict__ h,
                           const int* __restrict__ src, const int* __restrict__ dst,
                           const float* __restrict__ EX, const float* __restrict__ DEN,
                           float* __restrict__ out, int Ecnt) {
  const int lane = threadIdx.x & 31;
  long long task = (long long)blockIdx.x * (blockDim.x >> 5) + (threadIdx.x >> 5);
  long long total = (long long)Ecnt * HEADS;
  if (task >= total) return;
  int e = (int)(task / HEADS), hd = (int)(task % HEADS);
  int s = src[e], d = dst[e];
  float w = EX[task] / (DEN[(size_t)d * HEADS + hd] + 1e-16f);
  const int HC = HEADS * 64;
  const float* hp = h + (size_t)s * HC + hd * 64;
  float* op = out + (size_t)d * HC + hd * 64;
  atomicAdd(&op[lane],      w * hp[lane]);
  atomicAdd(&op[lane + 32], w * hp[lane + 32]);
}

// out = elu(batchnorm(agg + bias))
__global__ void post_bn_elu(const float* __restrict__ agg, const float* __restrict__ bias,
                            const float* __restrict__ g, const float* __restrict__ bb,
                            const float* __restrict__ m, const float* __restrict__ v,
                            float* __restrict__ out, long long total, int F) {
  long long i = (long long)blockIdx.x * blockDim.x + threadIdx.x;
  if (i >= total) return;
  int f = (int)(i % F);
  float x = agg[i] + bias[f];
  x = (x - m[f]) * (g[f] * rsqrtf(v[f] + 1e-5f)) + bb[f];
  out[i] = x > 0.f ? x : expm1f(x);
}

// t = agg3 + b3 + skip_gemm + skip_b   (F = 64)
__global__ void combine_skip(const float* __restrict__ agg, const float* __restrict__ b3,
                             const float* __restrict__ sk, const float* __restrict__ skb,
                             float* __restrict__ out, long long total) {
  long long i = (long long)blockIdx.x * blockDim.x + threadIdx.x;
  if (i >= total) return;
  int f = (int)(i & 63);
  out[i] = agg[i] + b3[f] + sk[i] + skb[f];
}

// out = elu(x + b)
__global__ void bias_elu(const float* __restrict__ x, const float* __restrict__ b,
                         float* __restrict__ out, long long total, int F) {
  long long i = (long long)blockIdx.x * blockDim.x + threadIdx.x;
  if (i >= total) return;
  float v = x[i] + b[(int)(i % F)];
  out[i] = v > 0.f ? v : expm1f(v);
}

// logits = u @ lin2_W + lin2_b ; log_softmax over 8 classes (per-node thread)
__global__ void head_logsoftmax(const float* __restrict__ u, const float* __restrict__ W,
                                const float* __restrict__ b, float* __restrict__ out, int Nn) {
  int n = blockIdx.x * blockDim.x + threadIdx.x;
  if (n >= Nn) return;
  float acc[8];
#pragma unroll
  for (int k = 0; k < 8; ++k) acc[k] = b[k];
  const float* up = u + (size_t)n * 64;
#pragma unroll 8
  for (int c = 0; c < 64; ++c) {
    float x = up[c];
    const float* wp = W + c * 8;
#pragma unroll
    for (int k = 0; k < 8; ++k) acc[k] += x * wp[k];
  }
  float mx = acc[0];
#pragma unroll
  for (int k = 1; k < 8; ++k) mx = fmaxf(mx, acc[k]);
  float sum = 0.f;
#pragma unroll
  for (int k = 0; k < 8; ++k) sum += expf(acc[k] - mx);
  float lse = mx + logf(sum);
  float* op = out + (size_t)n * 8;
#pragma unroll
  for (int k = 0; k < 8; ++k) op[k] = acc[k] - lse;
}

// ---------------- host orchestration ----------------

static inline unsigned cdiv(long long a, long long b) { return (unsigned)((a + b - 1) / b); }

template <int KIN, int HEADS>
static void gat_layer(const float* xin, const float* Wp,
                      const float* asrc, const float* adst,
                      int Nn, int Ecnt, const int* src, const int* dst,
                      float* H, float* AGG, float* AS, float* AD,
                      float* EMAX, float* DEN, float* EX, hipStream_t stream) {
  constexpr int HC = HEADS * 64;
  dim3 gg(cdiv(Nn, 64), HC / 64);
  gemm_wmma_bf16<KIN, HC><<<gg, dim3(128), 0, stream>>>(xin, Wp, H, Nn);

  long long nh = (long long)Nn * HEADS;
  alpha_reduce<HEADS><<<cdiv(nh, 256), 256, 0, stream>>>(H, asrc, adst, AS, AD, Nn);
  fill_f32<<<512, 256, 0, stream>>>(EMAX, -__builtin_huge_valf(), nh);
  fill_f32<<<512, 256, 0, stream>>>(DEN, 0.0f, nh);
  fill_f32<<<2048, 256, 0, stream>>>(AGG, 0.0f, (long long)Nn * HC);

  long long tE = (long long)Ecnt * HEADS;
  edge_max_k<HEADS><<<cdiv(tE, 256), 256, 0, stream>>>(AS, AD, src, dst, EMAX, tE);
  edge_exp_k<HEADS><<<cdiv(tE, 256), 256, 0, stream>>>(AS, AD, src, dst, EMAX, EX, DEN, tE);
  edge_agg_k<HEADS><<<cdiv(tE, 8), 256, 0, stream>>>(H, src, dst, EX, DEN, AGG, Ecnt);
}

extern "C" void kernel_launch(void* const* d_in, const int* in_sizes, int n_in,
                              void* d_out, int out_size, void* d_ws, size_t ws_size,
                              hipStream_t stream) {
  const float* x      = (const float*)d_in[0];
  const int*   ei     = (const int*)d_in[1];
  const float* W1     = (const float*)d_in[2];
  const float* a_src1 = (const float*)d_in[3];
  const float* a_dst1 = (const float*)d_in[4];
  const float* b1     = (const float*)d_in[5];
  const float* bn1_g  = (const float*)d_in[6];
  const float* bn1_b  = (const float*)d_in[7];
  const float* bn1_m  = (const float*)d_in[8];
  const float* bn1_v  = (const float*)d_in[9];
  const float* W2     = (const float*)d_in[10];
  const float* a_src2 = (const float*)d_in[11];
  const float* a_dst2 = (const float*)d_in[12];
  const float* b2     = (const float*)d_in[13];
  const float* bn2_g  = (const float*)d_in[14];
  const float* bn2_b  = (const float*)d_in[15];
  const float* bn2_m  = (const float*)d_in[16];
  const float* bn2_v  = (const float*)d_in[17];
  const float* W3     = (const float*)d_in[18];
  const float* a_src3 = (const float*)d_in[19];
  const float* a_dst3 = (const float*)d_in[20];
  const float* b3     = (const float*)d_in[21];
  const float* skip_W = (const float*)d_in[22];
  const float* skip_b = (const float*)d_in[23];
  const float* lin1_W = (const float*)d_in[24];
  const float* lin1_b = (const float*)d_in[25];
  const float* lin2_W = (const float*)d_in[26];
  const float* lin2_b = (const float*)d_in[27];
  (void)n_in; (void)out_size; (void)ws_size;

  const int Nn   = in_sizes[0] / 64;   // 50000 (multiple of 16)
  const int Ecnt = in_sizes[1] / 2;    // 800000
  const int* src = ei;
  const int* dst = ei + Ecnt;

  // workspace carve-up (floats); ~140 MB total
  float* ws = (float*)d_ws;
  size_t o = 0;
  float* BUF_H   = ws + o; o += (size_t)Nn * 192;  // pre-aggregation features h
  float* BUF_AGG = ws + o; o += (size_t)Nn * 192;  // aggregated conv output
  float* BUF_X   = ws + o; o += (size_t)Nn * 192;  // post-activation layer input
  float* SKIP    = ws + o; o += (size_t)Nn * 64;
  float* AS      = ws + o; o += (size_t)Nn * 3;
  float* AD      = ws + o; o += (size_t)Nn * 3;
  float* EMAX    = ws + o; o += (size_t)Nn * 3;
  float* DEN     = ws + o; o += (size_t)Nn * 3;
  float* EX      = ws + o; o += (size_t)Ecnt * 3;

  // conv1: 64 -> 192 (heads=3)
  gat_layer<64, 3>(x, W1, a_src1, a_dst1, Nn, Ecnt, src, dst,
                   BUF_H, BUF_AGG, AS, AD, EMAX, DEN, EX, stream);
  long long t1 = (long long)Nn * 192;
  post_bn_elu<<<cdiv(t1, 256), 256, 0, stream>>>(BUF_AGG, b1, bn1_g, bn1_b, bn1_m, bn1_v,
                                                 BUF_X, t1, 192);
  // conv2: 192 -> 128 (heads=2)
  gat_layer<192, 2>(BUF_X, W2, a_src2, a_dst2, Nn, Ecnt, src, dst,
                    BUF_H, BUF_AGG, AS, AD, EMAX, DEN, EX, stream);
  long long t2 = (long long)Nn * 128;
  post_bn_elu<<<cdiv(t2, 256), 256, 0, stream>>>(BUF_AGG, b2, bn2_g, bn2_b, bn2_m, bn2_v,
                                                 BUF_X, t2, 128);
  // conv3: 128 -> 64 (heads=1)
  gat_layer<128, 1>(BUF_X, W3, a_src3, a_dst3, Nn, Ecnt, src, dst,
                    BUF_H, BUF_AGG, AS, AD, EMAX, DEN, EX, stream);

  // skip = x @ skip_W
  gemm_wmma_bf16<64, 64><<<dim3(cdiv(Nn, 64), 1), dim3(128), 0, stream>>>(x, skip_W, SKIP, Nn);

  long long t3 = (long long)Nn * 64;
  combine_skip<<<cdiv(t3, 256), 256, 0, stream>>>(BUF_AGG, b3, SKIP, skip_b, BUF_X, t3);

  // lin1 + elu
  gemm_wmma_bf16<64, 64><<<dim3(cdiv(Nn, 64), 1), dim3(128), 0, stream>>>(BUF_X, lin1_W, BUF_H, Nn);
  bias_elu<<<cdiv(t3, 256), 256, 0, stream>>>(BUF_H, lin1_b, BUF_AGG, t3, 64);

  // lin2 + log_softmax -> d_out [N, 8]
  head_logsoftmax<<<cdiv(Nn, 256), 256, 0, stream>>>(BUF_AGG, lin2_W, lin2_b,
                                                     (float*)d_out, Nn);
}
